// teacherGRU_30477087932534
// MI455X (gfx1250) — compile-verified, matrix operand order checked by reading, not run on hardware
//
#include <hip/hip_runtime.h>

typedef __bf16 bf16;
typedef __attribute__((ext_vector_type(16))) __bf16 v16bf;
typedef __attribute__((ext_vector_type(8)))  __bf16 v8bf;
typedef __attribute__((ext_vector_type(8)))  float  v8f;

constexpr int Bn = 128;   // batch
constexpr int Sn = 256;   // seq len
constexpr int In = 512;   // input dim
constexpr int On = 512;   // output dim
constexpr int Hn = 1024;  // hidden
constexpr int IOn = In + On;      // 1024
constexpr int G3 = 3 * Hn;        // 3072
constexpr int Mrows = Sn * Bn;    // 32768 rows of the big GEMMs
constexpr int KC = 512;           // K staging chunk (bf16 elements)

// -------- workspace layout (bytes) --------
constexpr size_t OFF_WIH    = 0;                                  // bf16 [3H, IO]
constexpr size_t OFF_WHH    = OFF_WIH + (size_t)G3 * IOn * 2;     // bf16 [3H, H]
constexpr size_t OFF_LW     = OFF_WHH + (size_t)G3 * Hn * 2;      // bf16 [O, H]
constexpr size_t OFF_X      = OFF_LW  + (size_t)On * Hn * 2;      // bf16 [S*B, IO]
constexpr size_t OFF_GI     = OFF_X   + (size_t)Mrows * IOn * 2;  // f32  [S*B, 3H]
constexpr size_t OFF_HBUF   = OFF_GI  + (size_t)Mrows * G3 * 4;   // bf16 [(S+1), B, H]
constexpr size_t OFF_LOGITS = OFF_HBUF + (size_t)(Sn + 1) * Bn * Hn * 2; // f32 [S*B, O]

// ---------------------------------------------------------------------------
// bf16 WMMA fragment loader (CDNA5 16-bit operand layout):
// lane l -> row/col = l&15, k-group base kg = (l>>4)*8;
// lane holds k = kg..kg+7 and kg+16..kg+23 -> two contiguous 16B vectors.
// p must point at &Mat[row][k0 + kg] of a row-major matrix (16B aligned).
// ---------------------------------------------------------------------------
__device__ __forceinline__ v16bf load_frag16(const bf16* p) {
    v8bf lo = *(const v8bf*)p;
    v8bf hi = *(const v8bf*)(p + 16);
    return __builtin_shufflevector(lo, hi, 0, 1, 2, 3, 4, 5, 6, 7,
                                           8, 9, 10, 11, 12, 13, 14, 15);
}

__device__ __forceinline__ v8f wmma_bf16(v16bf a, v16bf b, v8f c) {
    return __builtin_amdgcn_wmma_f32_16x16x32_bf16(false, a, false, b, (short)0, c,
                                                   false, false);
}

// Async global->LDS copy of one 16-byte chunk (gfx1250 async path, ASYNCcnt).
// vdst = 32-bit LDS byte address, vaddr = 64-bit global address.
__device__ __forceinline__ void async_copy16(const bf16* g, bf16* l) {
    unsigned lds_off = (unsigned)(unsigned long long)l;   // low 32 bits = LDS offset
    unsigned long long ga = (unsigned long long)g;
    asm volatile("global_load_async_to_lds_b128 %0, %1, off"
                 :: "v"(lds_off), "v"(ga) : "memory");
}

__device__ __forceinline__ void async_wait_and_barrier() {
    asm volatile("s_wait_asynccnt 0x0" ::: "memory");
    __syncthreads();
}

__device__ __forceinline__ float sigmoidf_(float x) {
    return 1.0f / (1.0f + __expf(-x));
}

// ---------------------------------------------------------------------------
// 0) fp32 -> bf16 weight conversion (W_ih, W_hh, lin_w)
// ---------------------------------------------------------------------------
__global__ void k_prep(const float* __restrict__ wih, const float* __restrict__ whh,
                       const float* __restrict__ lw,
                       bf16* __restrict__ wih_b, bf16* __restrict__ whh_b,
                       bf16* __restrict__ lw_b) {
    long i = (long)blockIdx.x * 256 + threadIdx.x;
    const long n1 = (long)G3 * IOn;
    const long n2 = n1 + (long)G3 * Hn;
    const long n3 = n2 + (long)On * Hn;
    if (i < n1)      wih_b[i]      = (bf16)wih[i];
    else if (i < n2) whh_b[i - n1] = (bf16)whh[i - n1];
    else if (i < n3) lw_b[i - n2]  = (bf16)lw[i - n2];
}

// ---------------------------------------------------------------------------
// 1) build X[s*B+b][0:I]=y[b,s], X[..][I:]= (s==0? gotoken : gt[b,s-1]) as bf16
// ---------------------------------------------------------------------------
__global__ void k_build_x(const float* __restrict__ y, const float* __restrict__ gt,
                          const float* __restrict__ gotok, bf16* __restrict__ X) {
    long i = (long)blockIdx.x * 256 + threadIdx.x;
    if (i >= (long)Mrows * IOn) return;
    int c = (int)(i % IOn);
    long r = i / IOn;
    int b = (int)(r % Bn);
    int s = (int)(r / Bn);
    float v;
    if (c < In) {
        v = y[((long)b * Sn + s) * In + c];
    } else {
        int cc = c - In;
        v = (s == 0) ? gotok[cc] : gt[((long)b * Sn + (s - 1)) * On + cc];
    }
    X[i] = (bf16)v;
}

__global__ void k_init_h0(const float* __restrict__ hx, bf16* __restrict__ h0) {
    int i = blockIdx.x * 256 + threadIdx.x;
    if (i < Bn * Hn) h0[i] = (bf16)hx[i];
}

// ---------------------------------------------------------------------------
// 2) gi = X @ W_ih^T + bias_ih  (M=32768, N=3072, K=1024)
// Block: 8 waves x one 16-row M-tile each, all sharing a 64-col weight panel
// staged to LDS in two 512-K chunks via async global->LDS loads.
// grid: (48 n-groups, 256 m-blocks) x 256 threads
// ---------------------------------------------------------------------------
__global__ void k_gemm_gi(const bf16* __restrict__ X, const bf16* __restrict__ W,
                          const float* __restrict__ bias, float* __restrict__ gi) {
    __shared__ bf16 ldsB[64 * KC];               // 64 KB
    int lane  = threadIdx.x & 31;
    int wave  = threadIdx.x >> 5;
    int mbase = (blockIdx.y * 8 + wave) * 16;
    int nbase = blockIdx.x * 64;
    int nl    = lane & 15;
    int kg    = (lane >> 4) << 3;

    v8f acc[4];
#pragma unroll
    for (int t = 0; t < 4; ++t) {
        float bv = bias[nbase + t * 16 + nl];
#pragma unroll
        for (int e = 0; e < 8; ++e) acc[t][e] = bv;
    }

    const bf16* arow = X + (long)(mbase + nl) * IOn + kg;

    for (int kc = 0; kc < IOn; kc += KC) {
        // stage 64 rows x KC cols of W into LDS (async, 16B chunks)
        for (int i = threadIdx.x; i < 64 * (KC / 8); i += 256) {
            int r = i >> 6;                       // KC/8 == 64 chunks per row
            int c = i & 63;
            async_copy16(W + (long)(nbase + r) * IOn + kc + c * 8,
                         ldsB + r * KC + c * 8);
        }
        async_wait_and_barrier();

        for (int k0 = 0; k0 < KC; k0 += 32) {
            // issue ALL fragment loads for this k-step first, then all WMMAs,
            // so one dscnt wait covers four matrix ops.
            v16bf a = load_frag16(arow + kc + k0);
            v16bf bf0 = load_frag16(ldsB + (0 * 16 + nl) * KC + k0 + kg);
            v16bf bf1 = load_frag16(ldsB + (1 * 16 + nl) * KC + k0 + kg);
            v16bf bf2 = load_frag16(ldsB + (2 * 16 + nl) * KC + k0 + kg);
            v16bf bf3 = load_frag16(ldsB + (3 * 16 + nl) * KC + k0 + kg);
            acc[0] = wmma_bf16(a, bf0, acc[0]);
            acc[1] = wmma_bf16(a, bf1, acc[1]);
            acc[2] = wmma_bf16(a, bf2, acc[2]);
            acc[3] = wmma_bf16(a, bf3, acc[3]);
        }
        __syncthreads();                          // before next chunk overwrites LDS
    }

#pragma unroll
    for (int t = 0; t < 4; ++t) {
        int g = nbase + t * 16 + nl;
#pragma unroll
        for (int e = 0; e < 8; ++e) {
            int m = mbase + e + kg;               // C-layout: M = e + 8*(lane>=16)
            gi[(long)m * G3 + g] = acc[t][e];
        }
    }
}

// ---------------------------------------------------------------------------
// 3) one GRU timestep: gh = h@W_hh^T + b_hh, fused gates.
// Block: 8 waves = 8 m-tiles sharing one 16-col tile of all three gate
// weight panels (48 rows), staged to LDS via async loads in 512-K chunks.
// grid: 64 n-tiles x 256 threads. Launched S times.
// ---------------------------------------------------------------------------
__global__ void k_gru_step(const bf16* __restrict__ Whh, const float* __restrict__ bhh,
                           const float* __restrict__ gi, const bf16* __restrict__ hprev,
                           bf16* __restrict__ hnext, float* __restrict__ preact,
                           float* __restrict__ hx_final, int t) {
    __shared__ bf16 ldsB[48 * KC];               // 48 KB: [gate][16 rows][KC]
    int lane  = threadIdx.x & 31;
    int mbase = (threadIdx.x >> 5) * 16;         // m-tile = wave id
    int nbase = blockIdx.x * 16;                 // n-tile (within H)
    int nl    = lane & 15;
    int kg    = (lane >> 4) << 3;
    int hcol  = nbase + nl;

    v8f accR, accZ, accN;
    {
        float br = bhh[hcol], bz = bhh[Hn + hcol], bn = bhh[2 * Hn + hcol];
#pragma unroll
        for (int e = 0; e < 8; ++e) { accR[e] = br; accZ[e] = bz; accN[e] = bn; }
    }

    const bf16* arow = hprev + (long)(mbase + nl) * Hn + kg;

    for (int kc = 0; kc < Hn; kc += KC) {
        // stage 3 gates x 16 rows x KC cols of W_hh into LDS
        for (int i = threadIdx.x; i < 48 * (KC / 8); i += 256) {
            int r = i >> 6;                       // 0..47
            int c = i & 63;
            int gate = r >> 4, rr = r & 15;
            async_copy16(Whh + ((long)gate * Hn + nbase + rr) * Hn + kc + c * 8,
                         ldsB + r * KC + c * 8);
        }
        async_wait_and_barrier();

        const bf16* lR = ldsB + nl * KC + kg;
        const bf16* lZ = lR + 16 * KC;
        const bf16* lN = lR + 32 * KC;
        for (int k0 = 0; k0 < KC; k0 += 32) {
            v16bf a  = load_frag16(arow + kc + k0);
            v16bf bR = load_frag16(lR + k0);
            v16bf bZ = load_frag16(lZ + k0);
            v16bf bN = load_frag16(lN + k0);
            accR = wmma_bf16(a, bR, accR);
            accZ = wmma_bf16(a, bZ, accZ);
            accN = wmma_bf16(a, bN, accN);
        }
        __syncthreads();
    }

#pragma unroll
    for (int e = 0; e < 8; ++e) {
        int b = mbase + e + kg;
        long grow = ((long)t * Bn + b) * G3;
        float ir  = gi[grow + hcol];
        float iz  = gi[grow + Hn + hcol];
        float inn = gi[grow + 2 * Hn + hcol];
        float rg  = sigmoidf_(ir + accR[e]);
        float zg  = sigmoidf_(iz + accZ[e]);
        float pre = inn + rg * accN[e];
        float ng  = tanhf(pre);
        float hp  = (float)hprev[(long)b * Hn + hcol];
        float hy  = ng + zg * (hp - ng);
        hnext[(long)b * Hn + hcol] = (bf16)hy;
        preact[((long)b * Sn + t) * Hn + hcol] = pre;
        if (t == Sn - 1) hx_final[(long)b * Hn + hcol] = hy;
    }
}

// ---------------------------------------------------------------------------
// 4) logits = hs @ lin_w^T + lin_b  (M=32768, N=512, K=1024)
// Same LDS-staged structure as k_gemm_gi. grid: (8, 256) x 256 threads
// ---------------------------------------------------------------------------
__global__ void k_gemm_logits(const bf16* __restrict__ Hs, const bf16* __restrict__ Lw,
                              const float* __restrict__ lb, float* __restrict__ logits) {
    __shared__ bf16 ldsB[64 * KC];               // 64 KB
    int lane  = threadIdx.x & 31;
    int wave  = threadIdx.x >> 5;
    int mbase = (blockIdx.y * 8 + wave) * 16;
    int nbase = blockIdx.x * 64;
    int nl    = lane & 15;
    int kg    = (lane >> 4) << 3;

    v8f acc[4];
#pragma unroll
    for (int t = 0; t < 4; ++t) {
        float bv = lb[nbase + t * 16 + nl];
#pragma unroll
        for (int e = 0; e < 8; ++e) acc[t][e] = bv;
    }

    const bf16* arow = Hs + (long)(mbase + nl) * Hn + kg;

    for (int kc = 0; kc < Hn; kc += KC) {
        for (int i = threadIdx.x; i < 64 * (KC / 8); i += 256) {
            int r = i >> 6;
            int c = i & 63;
            async_copy16(Lw + (long)(nbase + r) * Hn + kc + c * 8,
                         ldsB + r * KC + c * 8);
        }
        async_wait_and_barrier();

        for (int k0 = 0; k0 < KC; k0 += 32) {
            v16bf a = load_frag16(arow + kc + k0);
            v16bf bf0 = load_frag16(ldsB + (0 * 16 + nl) * KC + k0 + kg);
            v16bf bf1 = load_frag16(ldsB + (1 * 16 + nl) * KC + k0 + kg);
            v16bf bf2 = load_frag16(ldsB + (2 * 16 + nl) * KC + k0 + kg);
            v16bf bf3 = load_frag16(ldsB + (3 * 16 + nl) * KC + k0 + kg);
            acc[0] = wmma_bf16(a, bf0, acc[0]);
            acc[1] = wmma_bf16(a, bf1, acc[1]);
            acc[2] = wmma_bf16(a, bf2, acc[2]);
            acc[3] = wmma_bf16(a, bf3, acc[3]);
        }
        __syncthreads();
    }

#pragma unroll
    for (int t = 0; t < 4; ++t) {
        int o = nbase + t * 16 + nl;
#pragma unroll
        for (int e = 0; e < 8; ++e) {
            int m = mbase + e + kg;
            logits[(long)m * On + o] = acc[t][e];
        }
    }
}

// ---------------------------------------------------------------------------
// 5) softmax -> probs, argmax -> one-hot sampled.  one block per (s,b) row.
// ---------------------------------------------------------------------------
__global__ void k_softmax_sample(const float* __restrict__ logits,
                                 float* __restrict__ probs,
                                 float* __restrict__ sampled) {
    int row = blockIdx.x;              // row = s*B + b
    int s = row / Bn, b = row % Bn;
    const float* L = logits + (long)row * On;
    __shared__ float sv[256];
    __shared__ int   si[256];
    int tid = threadIdx.x;

    float m = -3.402823466e38f; int mi = On;
    for (int o = tid; o < On; o += 256) {
        float v = L[o];
        if (v > m) { m = v; mi = o; }
    }
    sv[tid] = m; si[tid] = mi;
    __syncthreads();
    for (int off = 128; off; off >>= 1) {
        if (tid < off) {
            float v2 = sv[tid + off]; int i2 = si[tid + off];
            if (v2 > sv[tid] || (v2 == sv[tid] && i2 < si[tid])) { sv[tid] = v2; si[tid] = i2; }
        }
        __syncthreads();
    }
    float rowmax = sv[0];
    int   rowarg = si[0];
    __syncthreads();

    float a = 0.0f;
    for (int o = tid; o < On; o += 256) a += __expf(L[o] - rowmax);
    sv[tid] = a;
    __syncthreads();
    for (int off = 128; off; off >>= 1) {
        if (tid < off) sv[tid] += sv[tid + off];
        __syncthreads();
    }
    float inv = 1.0f / sv[0];

    long obase = ((long)b * Sn + s) * On;
    for (int o = tid; o < On; o += 256) {
        probs[obase + o]   = __expf(L[o] - rowmax) * inv;
        sampled[obase + o] = (o == rowarg) ? 1.0f : 0.0f;
    }
}

// ---------------------------------------------------------------------------
extern "C" void kernel_launch(void* const* d_in, const int* in_sizes, int n_in,
                              void* d_out, int out_size, void* d_ws, size_t ws_size,
                              hipStream_t stream) {
    const float* y     = (const float*)d_in[0];
    const float* gt    = (const float*)d_in[1];
    const float* hx    = (const float*)d_in[2];
    const float* wih   = (const float*)d_in[3];
    const float* bih   = (const float*)d_in[4];
    const float* whh   = (const float*)d_in[5];
    const float* bhh   = (const float*)d_in[6];
    const float* lw    = (const float*)d_in[7];
    const float* lb    = (const float*)d_in[8];
    const float* gotok = (const float*)d_in[9];

    float* out     = (float*)d_out;
    float* probs   = out;                                    // [B,S,O]
    float* preact  = out + (long)Bn * Sn * On;               // [B,S,H]
    float* sampled = preact + (long)Bn * Sn * Hn;            // [B,S,O]
    float* hxf     = sampled + (long)Bn * Sn * On;           // [B,H]

    char* ws = (char*)d_ws;
    bf16*  Wih_b  = (bf16*)(ws + OFF_WIH);
    bf16*  Whh_b  = (bf16*)(ws + OFF_WHH);
    bf16*  Lw_b   = (bf16*)(ws + OFF_LW);
    bf16*  X      = (bf16*)(ws + OFF_X);
    float* gi     = (float*)(ws + OFF_GI);
    bf16*  hbuf   = (bf16*)(ws + OFF_HBUF);   // [(S+1), B, H]
    float* logits = (float*)(ws + OFF_LOGITS);

    // 0) weights to bf16
    {
        long total = (long)G3 * IOn + (long)G3 * Hn + (long)On * Hn;
        int blocks = (int)((total + 255) / 256);
        k_prep<<<blocks, 256, 0, stream>>>(wih, whh, lw, Wih_b, Whh_b, Lw_b);
    }
    // 1) build input matrix + h0
    {
        long total = (long)Mrows * IOn;
        k_build_x<<<(int)((total + 255) / 256), 256, 0, stream>>>(y, gt, gotok, X);
        k_init_h0<<<(Bn * Hn + 255) / 256, 256, 0, stream>>>(hx, hbuf);
    }
    // 2) input-side GEMM (WMMA bf16, async-LDS staged weights)
    {
        dim3 grid(G3 / 64, Mrows / (16 * 8));   // (48, 256)
        k_gemm_gi<<<grid, 256, 0, stream>>>(X, Wih_b, bih, gi);
    }
    // 3) recurrent scan: one WMMA step kernel per timestep
    for (int t = 0; t < Sn; ++t) {
        const bf16* hp = hbuf + (long)t * Bn * Hn;
        bf16*       hn = hbuf + (long)(t + 1) * Bn * Hn;
        k_gru_step<<<Hn / 16, 256, 0, stream>>>(Whh_b, bhh, gi, hp, hn, preact, hxf, t);
    }
    // 4) logits GEMM (rows of hbuf[1..S], layout row = s*B+b)
    {
        const bf16* Hs = hbuf + (long)Bn * Hn;
        dim3 grid(On / 64, Mrows / (16 * 8));   // (8, 256)
        k_gemm_logits<<<grid, 256, 0, stream>>>(Hs, Lw_b, lb, logits);
    }
    // 5) softmax + argmax one-hot
    k_softmax_sample<<<Mrows, 256, 0, stream>>>(logits, probs, sampled);
}
